// AdditiveTokenMixer_89661737271892
// MI455X (gfx1250) — matrix operand order repaired and checked by reading, not compiled, not run on hardware
//
#include <hip/hip_runtime.h>
#include <hip/hip_bf16.h>
#include <math.h>

// ---------------- problem constants ----------------
constexpr int BB  = 2;     // batch
constexpr int CD  = 256;   // DIM
constexpr int HH  = 48;
constexpr int WWD = 48;
constexpr int HWN = HH * WWD;   // 2304
constexpr int LLN = HWN;
constexpr int DIN = 512;   // DI
constexpr int NSN = 16;
constexpr int DRN = 16;
constexpr int KD  = 4;     // scan directions

typedef __attribute__((ext_vector_type(16))) _Float16 v16h;
typedef __attribute__((ext_vector_type(8)))  _Float16 v8h;
typedef __attribute__((ext_vector_type(2)))  _Float16 v2h;
typedef __attribute__((ext_vector_type(8)))  float    v8f;

// =====================================================================
// WMMA GEMM, double-buffered LDS staging:
//   C[b] (MxN) = A (MxK) * B[b] (KxN), fp32 in/out, f16 MACs via
//   v_wmma_f32_16x16x32_f16.
//
// Block = 256 threads (8 waves) -> one 16(M) x 128(N) tile of C.
//   * A panel 16x32 staged once per block (8x wave reuse).
//   * B panel 32x128 staged transposed [n][k]; each lane's fragment is
//     two aligned ds_load_b128 of contiguous halves.
//   * Two LDS panels: next panel's global loads are issued before the
//     current panel's fragment reads + WMMA, so load latency hides
//     behind matrix work; staging stores + one barrier close the step.
//   * f16 pairs packed -> ds_store_b32 (no scalar b16 stores).
// Exact grids: every GEMM here has M % 16 == 0 and N % 128 == 0, so no
// M/N guards. K is a template constant (256 / 512 / 16); K=16 zero-pads
// branchlessly via clamped address + select.
// Lane layouts per CDNA5 ISA 7.12.2.
// =====================================================================
constexpr int TN = 128;   // N per block
constexpr int TK = 32;    // K per step

template<int KDIM>
__global__ void wmma_gemm_tile(const float* __restrict__ A, const float* __restrict__ B,
                               float* __restrict__ C, int N,
                               long strideB, long strideC) {
    __shared__ _Float16 ldsA[2][16 * TK];    // [buf][m][k]
    __shared__ _Float16 ldsB[2][TN * TK];    // [buf][n][k] (transposed)

    const int tid  = threadIdx.x;
    const int wave = tid >> 5;
    const int lane = tid & 31;
    const int half = lane >> 4;
    const int lr   = lane & 15;

    const int nBase = blockIdx.x * TN;
    const int mBase = blockIdx.y * 16;
    const float* Bp = B + (long)blockIdx.z * strideB;
    float*       Cp = C + (long)blockIdx.z * strideC;

    // staging assignments (fixed per thread)
    const int ar = (tid * 2) >> 5;          // A row
    const int ac = (tid * 2) & 31;          // A col (even)

    float aR[2];        // staged A elements
    float bR[16];       // staged B elements: 8 (kpair,n) pairs x 2 k's

    // ---- register-load of one K-panel (global, coalesced) ----
    auto load_panel = [&](int k0) {
#pragma unroll
        for (int j = 0; j < 2; ++j) {
            int kg = k0 + ac + j;
            int kc = kg < KDIM ? kg : KDIM - 1;            // clamped addr
            float v = A[(long)(mBase + ar) * KDIM + kc];   // always valid
            aR[j] = kg < KDIM ? v : 0.0f;
        }
#pragma unroll
        for (int j = 0; j < 8; ++j) {
            int p  = tid + j * 256;                        // 0..2047
            int n  = p & (TN - 1);                         // coalesced in n
            int kp = p >> 7;                               // 0..15
#pragma unroll
            for (int q = 0; q < 2; ++q) {
                int kg = k0 + 2 * kp + q;
                int kc = kg < KDIM ? kg : KDIM - 1;
                float v = Bp[(long)kc * N + nBase + n];
                bR[j * 2 + q] = kg < KDIM ? v : 0.0f;
            }
        }
    };
    // ---- convert + packed ds_store_b32 into panel `buf` ----
    auto store_panel = [&](int buf) {
        *reinterpret_cast<v2h*>(&ldsA[buf][ar * TK + ac]) =
            v2h{(_Float16)aR[0], (_Float16)aR[1]};
#pragma unroll
        for (int j = 0; j < 8; ++j) {
            int p  = tid + j * 256;
            int n  = p & (TN - 1);
            int kp = p >> 7;
            *reinterpret_cast<v2h*>(&ldsB[buf][n * TK + 2 * kp]) =
                v2h{(_Float16)bR[j * 2], (_Float16)bR[j * 2 + 1]};
        }
    };

    // prologue: stage panel 0
    load_panel(0);
    store_panel(0);
    __syncthreads();

    v8f acc = {};
    const int nb = wave * 16 + lr;          // this lane's C column in tile
    for (int k0 = 0; k0 < KDIM; k0 += TK) {
        const int cur = (k0 / TK) & 1;
        const bool hasNext = (k0 + TK) < KDIM;
        if (hasNext) load_panel(k0 + TK);   // global loads in flight

        // ---- A fragment: row lr, K = {half*8..+7, 16+half*8..+7} ----
        v8h a0 = *reinterpret_cast<const v8h*>(&ldsA[cur][lr * TK + half * 8]);
        v8h a1 = *reinterpret_cast<const v8h*>(&ldsA[cur][lr * TK + 16 + half * 8]);
        v16h af = __builtin_shufflevector(a0, a1,
                     0,1,2,3,4,5,6,7,8,9,10,11,12,13,14,15);
        // ---- B fragment: col nb, K = {half*8..+7, 16+half*8..+7} ----
        v8h b0 = *reinterpret_cast<const v8h*>(&ldsB[cur][nb * TK + half * 8]);
        v8h b1 = *reinterpret_cast<const v8h*>(&ldsB[cur][nb * TK + 16 + half * 8]);
        v16h bf = __builtin_shufflevector(b0, b1,
                     0,1,2,3,4,5,6,7,8,9,10,11,12,13,14,15);

        acc = __builtin_amdgcn_wmma_f32_16x16x32_f16(
            /*neg_a=*/false, af, /*neg_b=*/false, bf,
            /*c_mod=*/(short)0, acc, /*reuse_a=*/false, /*reuse_b=*/false);

        if (hasNext) store_panel(cur ^ 1);  // fill the other panel
        __syncthreads();
    }
    // ---- store: acc element rr -> row mBase + half*8 + rr, col nb ----
#pragma unroll
    for (int rr = 0; rr < 8; ++rr) {
        int m = mBase + half * 8 + rr;
        Cp[(long)m * N + nBase + nb] = acc[rr];
    }
}

static inline void launch_gemm(const float* A, const float* B, float* C,
                               int M, int N, int Kd, long sB, long sC, int nb,
                               hipStream_t s) {
    dim3 g(N / TN, M / 16, nb);
    if (Kd == 256)      wmma_gemm_tile<256><<<g, 256, 0, s>>>(A, B, C, N, sB, sC);
    else if (Kd == 512) wmma_gemm_tile<512><<<g, 256, 0, s>>>(A, B, C, N, sB, sC);
    else                wmma_gemm_tile<16 ><<<g, 256, 0, s>>>(A, B, C, N, sB, sC);
}

// =====================================================================
// Generic depthwise conv (square kernel) with fused bias / SiLU /
// residual-add / gating-mul.  Input channel layout: base + b*inBStride
// + c*HWN + hw.  Output contiguous (B,C,HWN).
// =====================================================================
__global__ void dwconv_kernel(const float* __restrict__ x, const float* __restrict__ w,
                              const float* __restrict__ bias,
                              const float* __restrict__ addsrc, long addStride,
                              const float* __restrict__ mulsrc, long mulStride,
                              float* __restrict__ out,
                              int C, int ks, int pad, int act, long inBStride) {
    long idx = (long)blockIdx.x * blockDim.x + threadIdx.x;
    long total = (long)BB * C * HWN;
    if (idx >= total) return;
    int l = (int)(idx % HWN);
    int c = (int)((idx / HWN) % C);
    int b = (int)(idx / ((long)HWN * C));
    int hh = l / WWD, ww = l % WWD;
    const float* xin = x + (long)b * inBStride + (long)c * HWN;
    const float* wc  = w + (long)c * ks * ks;
    float acc = bias ? bias[c] : 0.0f;
    for (int ky = 0; ky < ks; ++ky) {
        int ih = hh - pad + ky;
        if (ih < 0 || ih >= HH) continue;
        for (int kx = 0; kx < ks; ++kx) {
            int iw = ww - pad + kx;
            if (iw < 0 || iw >= WWD) continue;
            acc = fmaf(wc[ky * ks + kx], xin[ih * WWD + iw], acc);
        }
    }
    if (act == 1) acc = acc / (1.0f + __expf(-acc));          // SiLU
    if (addsrc) acc += addsrc[(long)b * addStride + (long)c * HWN + l];
    if (mulsrc) acc *= mulsrc[(long)b * mulStride + (long)c * HWN + l];
    out[((long)b * C + c) * HWN + l] = acc;
}

// ---------------- small elementwise helpers ----------------
__global__ void zero_kernel(float* p, long n) {
    long i = (long)blockIdx.x * blockDim.x + threadIdx.x;
    if (i < n) p[i] = 0.0f;
}

__global__ void clip_kernel(float* p, long n) {
    long i = (long)blockIdx.x * blockDim.x + threadIdx.x;
    if (i < n) p[i] = fminf(fmaxf(p[i], 0.0f), 6.0f);
}

// xs[b,k,d,l]: 4 directional re-orderings of xi (B,DI,L)
__global__ void build_xs_kernel(const float* __restrict__ xi, float* __restrict__ xs) {
    long idx = (long)blockIdx.x * blockDim.x + threadIdx.x;
    long total = (long)BB * KD * DIN * LLN;
    if (idx >= total) return;
    int l = (int)(idx % LLN);
    long t = idx / LLN;
    int d = (int)(t % DIN); t /= DIN;
    int k = (int)(t % KD);
    int b = (int)(t / KD);
    int src;
    if (k == 0)      src = l;
    else if (k == 1) src = (l % HH) * WWD + (l / HH);           // HW transpose
    else if (k == 2) src = LLN - 1 - l;                         // reverse
    else { int lrr = LLN - 1 - l; src = (lrr % HH) * WWD + (lrr / HH); }
    xs[idx] = xi[((long)b * DIN + d) * LLN + src];
}

// dts = softplus(dt_lin + dtb[k,d]), in place on (B,K,DI,L)
__global__ void softplus_bias_kernel(float* __restrict__ dts, const float* __restrict__ dtb) {
    long idx = (long)blockIdx.x * blockDim.x + threadIdx.x;
    long total = (long)BB * KD * DIN * LLN;
    if (idx >= total) return;
    long t = idx / LLN;
    int d = (int)(t % DIN);
    int k = (int)((t / DIN) % KD);
    float x = dts[idx] + dtb[k * DIN + d];
    dts[idx] = (x > 20.0f) ? x : __logf(1.0f + __expf(x));
}

// =====================================================================
// Selective scan: one thread per (b,k,d) channel, NS=16 states in regs.
// h = exp(dt*A)*h + dt*u*B ; y = sum_n h*C ; out += u*D, scattered to the
// direction-mapped spatial position with global_atomic_add_f32.
// =====================================================================
__global__ void scan_kernel(const float* __restrict__ xs, const float* __restrict__ xdbl,
                            const float* __restrict__ dts, const float* __restrict__ al,
                            const float* __restrict__ dp, float* __restrict__ yacc) {
    int idx = blockIdx.x * blockDim.x + threadIdx.x;
    if (idx >= BB * KD * DIN) return;
    int d = idx % DIN;
    int k = (idx / DIN) % KD;
    int b = idx / (DIN * KD);

    float An[NSN], h[NSN];
#pragma unroll
    for (int n = 0; n < NSN; ++n) {
        An[n] = -__expf(al[((long)k * DIN + d) * NSN + n]);
        h[n] = 0.0f;
    }
    const float* u_p  = xs  + ((long)(b * KD + k) * DIN + d) * LLN;
    const float* dt_p = dts + ((long)(b * KD + k) * DIN + d) * LLN;
    const float* B_p  = xdbl + ((long)(b * KD + k) * (DRN + 2 * NSN) + DRN) * LLN;
    const float* C_p  = B_p + (long)NSN * LLN;
    float Dd = dp[k * DIN + d];
    float* yb = yacc + ((long)b * DIN + d) * LLN;

    for (int l = 0; l < LLN; ++l) {
        float dt = dt_p[l];
        float u  = u_p[l];
        float du = dt * u;
        float y = 0.0f;
#pragma unroll
        for (int n = 0; n < NSN; ++n) {
            float Bn = B_p[(long)n * LLN + l];
            float Cn = C_p[(long)n * LLN + l];
            h[n] = __expf(dt * An[n]) * h[n] + du * Bn;
            y = fmaf(h[n], Cn, y);
        }
        float outv = y + u * Dd;
        int pos;
        if (k == 0)      pos = l;
        else if (k == 1) pos = (l % HH) * WWD + (l / HH);
        else if (k == 2) pos = LLN - 1 - l;
        else { int lrr = LLN - 1 - l; pos = (lrr % HH) * WWD + (lrr / HH); }
        atomicAdd(&yb[pos], outv);
    }
}

// =====================================================================
// LayerNorm over DI per token + SiLU(z) gate. One 256-thread block per
// token (b,l); each thread owns d and d+256. Writes yln in (B,DI,L).
// =====================================================================
__global__ void ln_gate_kernel(const float* __restrict__ yacc, const float* __restrict__ xz,
                               const float* __restrict__ lnw, const float* __restrict__ lnb,
                               float* __restrict__ yln) {
    __shared__ float red[256];
    int t = blockIdx.x;                    // 0 .. BB*LLN-1
    int b = t / LLN, l = t % LLN;
    int tid = threadIdx.x;
    int d0 = tid, d1 = tid + 256;
    float v0 = yacc[((long)b * DIN + d0) * LLN + l];
    float v1 = yacc[((long)b * DIN + d1) * LLN + l];

    red[tid] = v0 + v1;
    __syncthreads();
    for (int off = 128; off > 0; off >>= 1) {
        if (tid < off) red[tid] += red[tid + off];
        __syncthreads();
    }
    float mu = red[0] * (1.0f / DIN);
    __syncthreads();
    float e0 = v0 - mu, e1 = v1 - mu;
    red[tid] = e0 * e0 + e1 * e1;
    __syncthreads();
    for (int off = 128; off > 0; off >>= 1) {
        if (tid < off) red[tid] += red[tid + off];
        __syncthreads();
    }
    float rs = rsqrtf(red[0] * (1.0f / DIN) + 1e-5f);

    float z0 = xz[((long)b * 2 * DIN + DIN + d0) * LLN + l];
    float z1 = xz[((long)b * 2 * DIN + DIN + d1) * LLN + l];
    float g0 = z0 / (1.0f + __expf(-z0));
    float g1 = z1 / (1.0f + __expf(-z1));
    yln[((long)b * DIN + d0) * LLN + l] = (e0 * rs * lnw[d0] + lnb[d0]) * g0;
    yln[((long)b * DIN + d1) * LLN + l] = (e1 * rs * lnw[d1] + lnb[d1]) * g1;
}

// mean over L per (b,c); one block per (b,c)
__global__ void gmean_kernel(const float* __restrict__ g, float* __restrict__ gmean) {
    __shared__ float red[256];
    int bc = blockIdx.x;
    float s = 0.0f;
    for (int l = threadIdx.x; l < LLN; l += 256) s += g[(long)bc * LLN + l];
    red[threadIdx.x] = s;
    __syncthreads();
    for (int off = 128; off > 0; off >>= 1) {
        if (threadIdx.x < off) red[threadIdx.x] += red[threadIdx.x + off];
        __syncthreads();
    }
    if (threadIdx.x == 0) gmean[bc] = red[0] * (1.0f / LLN);
}

// y1[b,o] = relu(cbr_g[o] * sum_c cbr_w[o,c]*gmean[b,c] + cbr_b[o])
__global__ void cbr_kernel(const float* __restrict__ gmean, const float* __restrict__ cbrw,
                           const float* __restrict__ cbrg, const float* __restrict__ cbrb,
                           float* __restrict__ y1) {
    int idx = blockIdx.x * blockDim.x + threadIdx.x;
    if (idx >= BB * CD) return;
    int o = idx % CD, b = idx / CD;
    float s = 0.0f;
    for (int c = 0; c < CD; ++c) s = fmaf(cbrw[o * CD + c], gmean[b * CD + c], s);
    s = s * cbrg[o] + cbrb[o];
    y1[idx] = s > 0.0f ? s : 0.0f;
}

// out = g * (y1[b,c] + y2[b,c,l])
__global__ void final_kernel(const float* __restrict__ g, const float* __restrict__ y1,
                             const float* __restrict__ y2, float* __restrict__ out) {
    long idx = (long)blockIdx.x * blockDim.x + threadIdx.x;
    long total = (long)BB * CD * HWN;
    if (idx >= total) return;
    long bc = idx / HWN;
    out[idx] = g[idx] * (y1[bc] + y2[idx]);
}

// =====================================================================
// Host-side SS2D stage (WMMA GEMMs + scan + LN)
// =====================================================================
struct Scratch {
    float *xz, *xi, *xs, *xdbl, *dts, *yacc, *yln;
};

static inline int eblocks(long n) { return (int)((n + 255) / 256); }

static void run_ss2d(const float* xin,               // (B, CD, L) contiguous
                     const float* in_w, const float* cw, const float* cb,
                     const float* xp, const float* dtw, const float* dtb,
                     const float* al, const float* dp, const float* lnw,
                     const float* lnb, const float* ow,
                     float* outp,                     // (B, CD, L)
                     const Scratch& s, hipStream_t st) {
    // 1) in-projection: xz[b] (2DI x L) = in_w (2DI x CD) * xin[b]
    launch_gemm(in_w, xin, s.xz, 2 * DIN, LLN, CD,
                (long)CD * LLN, (long)2 * DIN * LLN, BB, st);
    // 2) 3x3 dwconv + SiLU on first DI channels of xz -> xi
    dwconv_kernel<<<eblocks((long)BB * DIN * HWN), 256, 0, st>>>(
        s.xz, cw, cb, nullptr, 0, nullptr, 0, s.xi, DIN, 3, 1, 1, (long)2 * DIN * LLN);
    // 3) 4 directional orderings
    build_xs_kernel<<<eblocks((long)BB * KD * DIN * LLN), 256, 0, st>>>(s.xi, s.xs);
    // 4) x_dbl: per direction k, (48 x L) = xp[k] (48 x DI) * xs[b,k]
    for (int k = 0; k < KD; ++k) {
        launch_gemm(xp + (long)k * 48 * DIN,
                    s.xs + (long)k * DIN * LLN,
                    s.xdbl + (long)k * 48 * LLN,
                    48, LLN, DIN,
                    (long)KD * DIN * LLN, (long)KD * 48 * LLN, BB, st);
    }
    // 5) dt projection: (DI x L) = dtw[k] (DI x DR) * x_dbl rows 0..DR-1 (K=16, zero padded)
    for (int k = 0; k < KD; ++k) {
        launch_gemm(dtw + (long)k * DIN * DRN,
                    s.xdbl + (long)k * 48 * LLN,
                    s.dts + (long)k * DIN * LLN,
                    DIN, LLN, DRN,
                    (long)KD * 48 * LLN, (long)KD * DIN * LLN, BB, st);
    }
    softplus_bias_kernel<<<eblocks((long)BB * KD * DIN * LLN), 256, 0, st>>>(s.dts, dtb);
    // 6) selective scan, atomically merging 4 directions into yacc
    zero_kernel<<<eblocks((long)BB * DIN * LLN), 256, 0, st>>>(s.yacc, (long)BB * DIN * LLN);
    scan_kernel<<<(BB * KD * DIN + 255) / 256, 256, 0, st>>>(s.xs, s.xdbl, s.dts, al, dp, s.yacc);
    // 7) LayerNorm over DI + SiLU(z) gate
    ln_gate_kernel<<<BB * LLN, 256, 0, st>>>(s.yacc, s.xz, lnw, lnb, s.yln);
    // 8) out projection: (CD x L) = ow (CD x DI) * yln[b]
    launch_gemm(ow, s.yln, outp, CD, LLN, DIN,
                (long)DIN * LLN, (long)CD * LLN, BB, st);
}

// =====================================================================
extern "C" void kernel_launch(void* const* d_in, const int* in_sizes, int n_in,
                              void* d_out, int out_size, void* d_ws, size_t ws_size,
                              hipStream_t stream) {
    // ---- inputs in setup_inputs() dict order ----
    const float* x       = (const float*)d_in[0];
    const float* replk_w = (const float*)d_in[1];
    const float* replk_b = (const float*)d_in[2];
    const float* qkv_w   = (const float*)d_in[3];
    const float* q_w     = (const float*)d_in[4];
    const float* q_b     = (const float*)d_in[5];
    const float* k_w     = (const float*)d_in[6];
    const float* k_b     = (const float*)d_in[7];
    const float* dwc_w   = (const float*)d_in[8];
    const float* dwc_b   = (const float*)d_in[9];
    const float* cbr_w   = (const float*)d_in[10];
    const float* cbr_g   = (const float*)d_in[11];
    const float* cbr_b   = (const float*)d_in[12];
    const float* s1p[11], *s2p[11];
    for (int i = 0; i < 11; ++i) s1p[i] = (const float*)d_in[13 + i];
    for (int i = 0; i < 11; ++i) s2p[i] = (const float*)d_in[24 + i];
    // order: in_w, cw, cb, xp, dtw, dtb, al, d, lnw, lnb, ow
    float* out = (float*)d_out;

    // ---- workspace carve-up ----
    float* ws = (float*)d_ws;
    long o = 0;
    float* buf_replk = ws + o; o += (long)BB * CD * LLN;
    Scratch sc;
    sc.xz   = ws + o; o += (long)BB * 2 * DIN * LLN;
    sc.xi   = ws + o; o += (long)BB * DIN * LLN;
    sc.xs   = ws + o; o += (long)BB * KD * DIN * LLN;
    sc.xdbl = ws + o; o += (long)BB * KD * 48 * LLN;
    sc.dts  = ws + o; o += (long)BB * KD * DIN * LLN;
    sc.yacc = ws + o; o += (long)BB * DIN * LLN;
    sc.yln  = ws + o; o += (long)BB * DIN * LLN;
    float* buf_ss1  = ws + o; o += (long)BB * CD * LLN;
    float* buf_qkv  = ws + o; o += (long)BB * 3 * CD * LLN;
    float* buf_qc   = ws + o; o += (long)BB * CD * LLN;
    float* buf_s    = ws + o; o += (long)BB * CD * LLN;
    float* buf_g    = ws + o; o += (long)BB * CD * LLN;
    float* buf_y2   = ws + o; o += (long)BB * CD * LLN;
    float* buf_gm   = ws + o; o += (long)BB * CD;
    float* buf_y1   = ws + o; o += (long)BB * CD;
    (void)ws_size; (void)in_sizes; (void)n_in; (void)out_size;

    // ---- 1. 13x13 depthwise "RepLK" conv ----
    dwconv_kernel<<<eblocks((long)BB * CD * HWN), 256, 0, stream>>>(
        x, replk_w, replk_b, nullptr, 0, nullptr, 0, buf_replk,
        CD, 13, 6, 0, (long)CD * LLN);

    // ---- 2. SS2D stage 1 ----
    run_ss2d(buf_replk, s1p[0], s1p[1], s1p[2], s1p[3], s1p[4], s1p[5],
             s1p[6], s1p[7], s1p[8], s1p[9], s1p[10], buf_ss1, sc, stream);

    // ---- 3. clip(0,6) then QKV projection (WMMA) ----
    clip_kernel<<<eblocks((long)BB * CD * LLN), 256, 0, stream>>>(buf_ss1, (long)BB * CD * LLN);
    launch_gemm(qkv_w, buf_ss1, buf_qkv, 3 * CD, LLN, CD,
                (long)CD * LLN, (long)3 * CD * LLN, BB, stream);

    // ---- 4. additive mixer: g = dwconv(dwconv(q)+dwconv(k)) * v ----
    const long qkvStride = (long)3 * CD * LLN;
    dwconv_kernel<<<eblocks((long)BB * CD * HWN), 256, 0, stream>>>(
        buf_qkv, q_w, q_b, nullptr, 0, nullptr, 0, buf_qc, CD, 3, 1, 0, qkvStride);
    dwconv_kernel<<<eblocks((long)BB * CD * HWN), 256, 0, stream>>>(
        buf_qkv + (long)CD * LLN, k_w, k_b, buf_qc, (long)CD * LLN, nullptr, 0,
        buf_s, CD, 3, 1, 0, qkvStride);
    dwconv_kernel<<<eblocks((long)BB * CD * HWN), 256, 0, stream>>>(
        buf_s, dwc_w, dwc_b, nullptr, 0, buf_qkv + (long)2 * CD * LLN, qkvStride,
        buf_g, CD, 3, 1, 0, (long)CD * LLN);

    // ---- 5. CBR branch on global-mean pooled g ----
    gmean_kernel<<<BB * CD, 256, 0, stream>>>(buf_g, buf_gm);
    cbr_kernel<<<(BB * CD + 255) / 256, 256, 0, stream>>>(buf_gm, cbr_w, cbr_g, cbr_b, buf_y1);

    // ---- 6. SS2D stage 2 on g ----
    run_ss2d(buf_g, s2p[0], s2p[1], s2p[2], s2p[3], s2p[4], s2p[5],
             s2p[6], s2p[7], s2p[8], s2p[9], s2p[10], buf_y2, sc, stream);

    // ---- 7. final gated combine ----
    final_kernel<<<eblocks((long)BB * CD * HWN), 256, 0, stream>>>(buf_g, buf_y1, buf_y2, out);
}
